// SpectralTCNVQVAE_58059367907960
// MI455X (gfx1250) — matrix operand order repaired and checked by reading, not compiled, not run on hardware
//
#include <hip/hip_runtime.h>
#include <hip/hip_bf16.h>
#include <cstdint>
#include <cstddef>

// ---------------------------------------------------------------------------
// Spectral TCN VQ-VAE for MI455X (gfx1250, wave32, WMMA f16 16x16x32).
// All 64-channel convs / codebook search run as WMMA GEMMs with f32 accum.
// Activations live in d_ws as f16 [pixel][row][64] with zeroed halo rows so
// every GEMM B-operand is a contiguous, 32B-aligned per-lane load.
// ---------------------------------------------------------------------------

typedef __attribute__((ext_vector_type(16))) _Float16 v16h;
typedef __attribute__((ext_vector_type(8)))  float    v8f;

#define NPIX 8192           // 2 * 64 * 64 spectra

// ---- workspace layout (bytes) ---------------------------------------------
static constexpr size_t SZ_Z   = (size_t)NPIX * 18 * 64 * 2;   // L=16 (+2 halo)
static constexpr size_t OFF_ZA = 0;
static constexpr size_t OFF_ZT = OFF_ZA + SZ_Z;
static constexpr size_t OFF_ZB = OFF_ZT + SZ_Z;
static constexpr size_t OFF_ZQ = OFF_ZB + SZ_Z;
static constexpr size_t SZ_A0  = (size_t)NPIX * 66 * 64 * 2;   // L=64 (+2 halo)
static constexpr size_t OFF_A0 = OFF_ZQ + SZ_Z;                // also reused as d2
static constexpr size_t SZ_A1  = (size_t)NPIX * 34 * 64 * 2;   // L=32 (+2 halo)
static constexpr size_t OFF_A1 = OFF_A0 + SZ_A0;               // also reused as d1
static constexpr size_t OFF_E16 = OFF_A1 + SZ_A1;              // f16 codebook 512x64
static constexpr size_t OFF_E2  = OFF_E16 + 512 * 64 * 2;      // |E|^2, 512 f32
static constexpr size_t WS_END  = OFF_E2 + 512 * 4;

// weight repack modes for the LDS A-matrix
#define WM_CONV    0   // W(Cout,Cin,K): A[co][tap*64+ci] = W[co][ci][tap]
#define WM_CONV1   1   // W(Cout,Cin,1)
#define WM_CT_EVEN 2   // convT even outputs: taps (3,1) of W(Cin,Cout,4)
#define WM_CT_ODD  3   // convT odd  outputs: taps (2,0)

// ---------------------------------------------------------------------------
__global__ void k_zero(uint4* __restrict__ p, size_t n16, float* __restrict__ diff) {
  size_t i = (size_t)blockIdx.x * blockDim.x + threadIdx.x;
  size_t stride = (size_t)gridDim.x * blockDim.x;
  uint4 z; z.x = z.y = z.z = z.w = 0u;
  for (; i < n16; i += stride) p[i] = z;
  if (blockIdx.x == 0 && threadIdx.x == 0) *diff = 0.0f;
}

// codebook -> f16 copy + |E_j|^2
__global__ void k_eprep(const float* __restrict__ E, _Float16* __restrict__ E16,
                        float* __restrict__ e2) {
  int t = blockIdx.x * 256 + threadIdx.x;
  if (t < 512 * 64) E16[t] = (_Float16)E[t];
  if (t < 512) {
    float s = 0.f;
    #pragma unroll 8
    for (int c = 0; c < 64; ++c) { float v = E[t * 64 + c]; s += v * v; }
    e2[t] = s;
  }
}

// enc_down0: Cin=1 conv (k=4, s=2, pad=1) + ReLU -> a0[pix][l+1][co]
__global__ void k_down0(const float* __restrict__ x, const float* __restrict__ W,
                        const float* __restrict__ B, _Float16* __restrict__ a0) {
  int t = blockIdx.x * 256 + threadIdx.x;        // NPIX*64*64 threads
  int pix = t >> 12, l = (t >> 6) & 63, co = t & 63;
  int b = pix >> 12, hw = pix & 4095;            // pix = (b*64+hh)*64+ww
  const float* xs = x + (size_t)b * 128 * 4096 + hw;   // x[b][band][hh][ww]
  float s = B[co];
  #pragma unroll
  for (int k = 0; k < 4; ++k) {
    int band = 2 * l + k - 1;
    if (band >= 0 && band < 128) s += W[co * 4 + k] * xs[(size_t)band * 4096];
  }
  s = s > 0.f ? s : 0.f;
  a0[((size_t)pix * 66 + (l + 1)) * 64 + co] = (_Float16)s;
}

// ---------------------------------------------------------------------------
// Generic conv-as-GEMM: D(64 x 16tok) = A(64 x KD) * B(KD x 16tok) + bias.
// B per token = KD contiguous f16 at in[pix][acoef*pos+boff][*].
// Store to out[pix][os*pos+ob][co] (opt ReLU / residual from res[pix][pos+1]).
// ---------------------------------------------------------------------------
template<int KD, int WMODE>
__global__ __launch_bounds__(256)
void conv_gemm(const _Float16* __restrict__ in, _Float16* __restrict__ out,
               const float* __restrict__ W, const float* __restrict__ Bi,
               const _Float16* __restrict__ res,
               int Lout, int Rin, int Rout,
               int acoef, int boff, int os, int ob,
               int in_relu, int out_relu) {
  __shared__ _Float16 sA[64 * KD];
  __shared__ float    sBias[64];

  for (int i = threadIdx.x; i < 64 * KD; i += 256) {
    int co = i / KD, idx = i - co * KD;
    int tap = idx >> 6, ci = idx & 63;
    float v;
    if (WMODE == WM_CONV)         v = W[(co * 64 + ci) * (KD / 64) + tap];
    else if (WMODE == WM_CONV1)   v = W[co * 64 + ci];
    else if (WMODE == WM_CT_EVEN) v = W[(ci * 64 + co) * 4 + (tap == 0 ? 3 : 1)];
    else                          v = W[(ci * 64 + co) * 4 + (tap == 0 ? 2 : 0)];
    sA[i] = (_Float16)v;
  }
  if (threadIdx.x < 64) sBias[threadIdx.x] = Bi[threadIdx.x];
  __syncthreads();

  const int lane = threadIdx.x & 31;
  const int wave = threadIdx.x >> 5;
  const int n15  = lane & 15;
  const int hi   = lane >> 4;                 // 0: lanes 0-15, 1: lanes 16-31
  const int ntile = (NPIX * Lout) >> 4;
  constexpr int NC = KD / 32;

  for (int tile = blockIdx.x * 8 + wave; tile < ntile; tile += gridDim.x * 8) {
    int tk  = tile * 16 + n15;                // this lane's token (column n)
    int pix = tk / Lout;
    int pos = tk - pix * Lout;

    // --- B operand: KD contiguous halves for this token ---
    const _Float16* bp = in + ((size_t)pix * Rin + (acoef * pos + boff)) * 64 + hi * 16;
    v16h bmat[NC];
    #pragma unroll
    for (int c = 0; c < NC; ++c) {
      bmat[c] = *(const v16h*)(bp + c * 32);
      if (in_relu) {
        #pragma unroll
        for (int h = 0; h < 16; ++h)
          bmat[c][h] = bmat[c][h] > (_Float16)0 ? bmat[c][h] : (_Float16)0;
      }
    }

    int orow = os * pos + ob;
    #pragma unroll
    for (int m = 0; m < 4; ++m) {
      const int co0 = m * 16 + hi * 8;        // first output channel this lane holds
      v8f acc;
      #pragma unroll
      for (int r = 0; r < 8; ++r) acc[r] = sBias[co0 + r];
      #pragma unroll
      for (int c = 0; c < NC; ++c) {
        const _Float16* ap = &sA[(m * 16 + n15) * KD + c * 32 + hi * 16];
        v16h a;
        #pragma unroll
        for (int h = 0; h < 8; ++h) { a[h] = ap[h]; a[h + 8] = ap[8 + h]; }
        acc = __builtin_amdgcn_wmma_f32_16x16x32_f16(false, a, false, bmat[c],
                                                     (short)0, acc, false, false);
      }
      _Float16* op = out + ((size_t)pix * Rout + orow) * 64 + co0;
      const _Float16* rp = res ? res + ((size_t)pix * Rin + (pos + 1)) * 64 + co0 : nullptr;
      #pragma unroll
      for (int r = 0; r < 8; ++r) {
        float v = acc[r];
        if (res) v += (float)rp[r];
        if (out_relu) v = v > 0.f ? v : 0.f;
        op[r] = (_Float16)v;
      }
    }
  }
}

// ---------------------------------------------------------------------------
// VQ: per 16-token tile, D = Z(16x64) @ E^T(64x512) via WMMA; argmin of
// |E|^2 - 2 z.E via branchless packed (score,code) u64 min; write ids (f32),
// zq (f16), accumulate commitment loss.
// ---------------------------------------------------------------------------
__global__ __launch_bounds__(256)
void k_vq(const _Float16* __restrict__ z, const _Float16* __restrict__ E16,
          const float* __restrict__ e2, _Float16* __restrict__ zq,
          float* __restrict__ ids, float* __restrict__ diff) {
  __shared__ float sE2[512];
  for (int i = threadIdx.x; i < 512; i += 256) sE2[i] = e2[i];
  __syncthreads();

  const int lane = threadIdx.x & 31;
  const int wave = threadIdx.x >> 5;
  const int n15  = lane & 15;
  const int hi   = lane >> 4;

  for (int tile = blockIdx.x * 8 + wave; tile < NPIX; tile += gridDim.x * 8) {
    // A operand: token row m = n15 of this tile
    int tk = tile * 16 + n15;
    int pix = tk >> 4, pos = tk & 15;
    const _Float16* zp = z + ((size_t)pix * 18 + pos + 1) * 64;
    v16h A0, A1;
    #pragma unroll
    for (int h = 0; h < 8; ++h) {
      A0[h]     = zp[hi * 16 + h];
      A0[h + 8] = zp[8 + hi * 16 + h];
      A1[h]     = zp[32 + hi * 16 + h];
      A1[h + 8] = zp[40 + hi * 16 + h];
    }

    // packed key: orderable(score) << 16 | code  -> min == (argmin, lowest id)
    unsigned long long mk[8];
    #pragma unroll
    for (int r = 0; r < 8; ++r) mk[r] = ~0ull;

    for (int j = 0; j < 32; ++j) {            // 32 code tiles of 16
      int code = j * 16 + n15;                // this lane's column
      const _Float16* ep = E16 + (size_t)code * 64 + hi * 16;
      v16h B0 = *(const v16h*)(ep);
      v16h B1 = *(const v16h*)(ep + 32);
      v8f acc = {};
      acc = __builtin_amdgcn_wmma_f32_16x16x32_f16(false, A0, false, B0, (short)0, acc, false, false);
      acc = __builtin_amdgcn_wmma_f32_16x16x32_f16(false, A1, false, B1, (short)0, acc, false, false);
      float bias = sE2[code];
      #pragma unroll
      for (int r = 0; r < 8; ++r) {
        float s = fmaf(-2.f, acc[r], bias);
        uint32_t fb = __float_as_uint(s);
        uint32_t ok = fb ^ (uint32_t)(((int32_t)fb >> 31) | 0x80000000);
        unsigned long long key = ((unsigned long long)ok << 16) | (unsigned)code;
        mk[r] = key < mk[r] ? key : mk[r];    // branchless v_cndmask pair
      }
    }
    // reduce across the 16 lanes that share the same token rows
    #pragma unroll
    for (int r = 0; r < 8; ++r) {
      #pragma unroll
      for (int off = 1; off < 16; off <<= 1) {
        unsigned long long o = __shfl_xor(mk[r], off, 32);
        mk[r] = o < mk[r] ? o : mk[r];
      }
    }
    int mini[8];
    #pragma unroll
    for (int r = 0; r < 8; ++r) mini[r] = (int)(mk[r] & 0xFFFFu);

    // every lane gets ids for all 16 tokens
    int id16[16];
    #pragma unroll
    for (int r = 0; r < 8; ++r) {
      int other = __shfl_xor(mini[r], 16, 32);
      id16[r]     = hi ? other   : mini[r];
      id16[r + 8] = hi ? mini[r] : other;
    }
    if (lane == 0) {
      #pragma unroll
      for (int m = 0; m < 16; ++m) ids[tile * 16 + m] = (float)id16[m];
    }
    // zq copy (straight-through value) + commitment loss partial
    float dsum = 0.f;
    #pragma unroll
    for (int m = 0; m < 16; ++m) {
      int tkm = tile * 16 + m;
      int pm = tkm >> 4, qm = tkm & 15;
      const _Float16* zr = z  + ((size_t)pm * 18 + qm + 1) * 64;
      _Float16*       qr = zq + ((size_t)pm * 18 + qm + 1) * 64;
      const _Float16* er = E16 + (size_t)id16[m] * 64;
      int c = lane * 2;
      float e0 = (float)er[c], e1 = (float)er[c + 1];
      float z0 = (float)zr[c], z1 = (float)zr[c + 1];
      qr[c] = (_Float16)e0; qr[c + 1] = (_Float16)e1;
      float d0 = e0 - z0, d1 = e1 - z1;
      dsum += d0 * d0 + d1 * d1;
    }
    #pragma unroll
    for (int off = 16; off >= 1; off >>= 1) dsum += __shfl_xor(dsum, off, 32);
    if (lane == 0) atomicAdd(diff, dsum);
  }
}

// last convT (64 -> 1, k4, s2, pad1) + recon scatter + diff finalize
__global__ void k_final(const _Float16* __restrict__ d2, const float* __restrict__ W,
                        const float* __restrict__ B, float* __restrict__ recon,
                        float* __restrict__ diff) {
  int t = blockIdx.x * 256 + threadIdx.x;     // band*NPIX + pix ; NPIX*128 threads
  if (t == 0) *diff = *diff * (1.0f / 8388608.0f);
  int pix = t & (NPIX - 1);
  int p   = t >> 13;                          // output band 0..127
  int q = p >> 1, par = p & 1;
  int r0 = q + par;                           // buffer row of first operand
  int k0 = par ? 2 : 3, k1 = par ? 0 : 1;
  const _Float16* a = d2 + ((size_t)pix * 66 + r0) * 64;
  const _Float16* b2 = a + 64;
  float s = B[0];
  #pragma unroll 8
  for (int ci = 0; ci < 64; ++ci)
    s += W[ci * 4 + k0] * (float)a[ci] + W[ci * 4 + k1] * (float)b2[ci];
  int bb = pix >> 12, hw = pix & 4095;
  recon[((size_t)bb * 128 + p) * 4096 + hw] = s;
}

// ---------------------------------------------------------------------------
extern "C" void kernel_launch(void* const* d_in, const int* in_sizes, int n_in,
                              void* d_out, int out_size, void* d_ws, size_t ws_size,
                              hipStream_t stream) {
  (void)in_sizes; (void)n_in; (void)out_size; (void)ws_size;
  // input order: x, enc_down(w,b)x3, enc_res(w3,b3,w1,b1)x4, codebook,
  //              dec_res(w3,b3,w1,b1)x4, dec_up(w,b)x3
  const float* x   = (const float*)d_in[0];
  const float* ew0 = (const float*)d_in[1];  const float* eb0 = (const float*)d_in[2];
  const float* ew1 = (const float*)d_in[3];  const float* eb1 = (const float*)d_in[4];
  const float* ew2 = (const float*)d_in[5];  const float* eb2 = (const float*)d_in[6];
  const float *erw3[4], *erb3[4], *erw1[4], *erb1[4];
  for (int i = 0; i < 4; ++i) {
    erw3[i] = (const float*)d_in[7 + 4 * i];  erb3[i] = (const float*)d_in[8 + 4 * i];
    erw1[i] = (const float*)d_in[9 + 4 * i];  erb1[i] = (const float*)d_in[10 + 4 * i];
  }
  const float* cb = (const float*)d_in[23];
  const float *drw3[4], *drb3[4], *drw1[4], *drb1[4];
  for (int i = 0; i < 4; ++i) {
    drw3[i] = (const float*)d_in[24 + 4 * i]; drb3[i] = (const float*)d_in[25 + 4 * i];
    drw1[i] = (const float*)d_in[26 + 4 * i]; drb1[i] = (const float*)d_in[27 + 4 * i];
  }
  const float* uw1 = (const float*)d_in[40]; const float* ub1 = (const float*)d_in[41];
  const float* uw2 = (const float*)d_in[42]; const float* ub2 = (const float*)d_in[43];
  const float* uw3 = (const float*)d_in[44]; const float* ub3 = (const float*)d_in[45];

  char* ws = (char*)d_ws;
  _Float16* zA  = (_Float16*)(ws + OFF_ZA);
  _Float16* zT  = (_Float16*)(ws + OFF_ZT);
  _Float16* zB  = (_Float16*)(ws + OFF_ZB);
  _Float16* zQ  = (_Float16*)(ws + OFF_ZQ);
  _Float16* a0  = (_Float16*)(ws + OFF_A0);   // later reused as d2 (R=66)
  _Float16* a1  = (_Float16*)(ws + OFF_A1);   // later reused as d1 (R=34)
  _Float16* E16 = (_Float16*)(ws + OFF_E16);
  float*    e2  = (float*)(ws + OFF_E2);

  float* out  = (float*)d_out;
  float* diff = out + 1048576;
  float* ids  = out + 1048577;

  k_zero<<<4096, 256, 0, stream>>>((uint4*)ws, WS_END / 16, diff);
  k_eprep<<<128, 256, 0, stream>>>(cb, E16, e2);
  k_down0<<<(NPIX * 4096) / 256, 256, 0, stream>>>(x, ew0, eb0, a0);

  // encoder downsampling (k4,s2): out = relu(conv(in))
  conv_gemm<256, WM_CONV><<<1024, 256, 0, stream>>>(a0, a1, ew1, eb1, nullptr,
                                                    32, 66, 34, 2, 0, 1, 1, 0, 1);
  conv_gemm<256, WM_CONV><<<1024, 256, 0, stream>>>(a1, zA, ew2, eb2, nullptr,
                                                    16, 34, 18, 2, 0, 1, 1, 0, 1);
  // encoder residual blocks: t = conv3(relu(x)); out = x + conv1(relu(t))
  {
    _Float16* X = zA; _Float16* Y = zB;
    for (int i = 0; i < 4; ++i) {
      conv_gemm<192, WM_CONV><<<1024, 256, 0, stream>>>(X, zT, erw3[i], erb3[i], nullptr,
                                                        16, 18, 18, 1, 0, 1, 1, 1, 0);
      conv_gemm<64, WM_CONV1><<<1024, 256, 0, stream>>>(zT, Y, erw1[i], erb1[i], X,
                                                        16, 18, 18, 1, 1, 1, 1, 1, 0);
      _Float16* tmp = X; X = Y; Y = tmp;
    }
    // final z in zA (4 swaps -> back in zA)
  }
  k_vq<<<1024, 256, 0, stream>>>(zA, E16, e2, zQ, ids, diff);

  // decoder residual blocks on zq (ping-pong zQ <-> zB; ends in zQ)
  {
    _Float16* X = zQ; _Float16* Y = zB;
    for (int i = 0; i < 4; ++i) {
      conv_gemm<192, WM_CONV><<<1024, 256, 0, stream>>>(X, zT, drw3[i], drb3[i], nullptr,
                                                        16, 18, 18, 1, 0, 1, 1, 1, 0);
      conv_gemm<64, WM_CONV1><<<1024, 256, 0, stream>>>(zT, Y, drw1[i], drb1[i], X,
                                                        16, 18, 18, 1, 1, 1, 1, 1, 0);
      _Float16* tmp = X; X = Y; Y = tmp;
    }
  }
  _Float16* d1 = a1;  // reuse (a1 consumed)
  _Float16* d2 = a0;  // reuse (a0 consumed)
  // convT1 (16->32), relu in (post-res), relu out; parity split
  conv_gemm<128, WM_CT_EVEN><<<1024, 256, 0, stream>>>(zQ, d1, uw1, ub1, nullptr,
                                                       16, 18, 34, 1, 0, 2, 1, 1, 1);
  conv_gemm<128, WM_CT_ODD><<<1024, 256, 0, stream>>>(zQ, d1, uw1, ub1, nullptr,
                                                      16, 18, 34, 1, 1, 2, 2, 1, 1);
  // convT2 (32->64), relu out
  conv_gemm<128, WM_CT_EVEN><<<1024, 256, 0, stream>>>(d1, d2, uw2, ub2, nullptr,
                                                       32, 34, 66, 1, 0, 2, 1, 0, 1);
  conv_gemm<128, WM_CT_ODD><<<1024, 256, 0, stream>>>(d1, d2, uw2, ub2, nullptr,
                                                      32, 34, 66, 1, 1, 2, 2, 0, 1);
  // convT3 (64ch -> 1 band x128) + recon layout + diff finalize
  k_final<<<(NPIX * 128) / 256, 256, 0, stream>>>(d2, uw3, ub3, out, diff);
}
